// BEVFuser_79937931313271
// MI455X (gfx1250) — matrix operand order repaired and checked by reading, not compiled, not run on hardware
//
#include <hip/hip_runtime.h>

// ---------------------------------------------------------------------------
// Types for CDNA5 WMMA (wave32): A/B = v16h (16 halfs = 8 VGPRs), C/D = v8f.
// ---------------------------------------------------------------------------
typedef __attribute__((ext_vector_type(16))) _Float16 v16h;
typedef __attribute__((ext_vector_type(8)))  _Float16 v8h;
typedef __attribute__((ext_vector_type(2)))  _Float16 v2h;
typedef __attribute__((ext_vector_type(8)))  float    v8f;

union V16 { v16h v; v8h h[2]; };

// Problem constants (fixed by the reference).
#define NH 4
#define NM 2
#define NP 4

// ---------------------------------------------------------------------------
// Generic WMMA GEMM:  D[M,N] = op( A[M,K](f16) @ Bp^T + bias (+ Cres) )
//   A      : row-major f16, leading dim lda (halfs)
//   Bp     : packed weights, N x Kp f16 (row n holds K contiguous), ldb = Kp
//   Block  : 256 threads = 8 waves; block tile 256(M) x 16*JT(N); K step 32.
//   Wave w : rows [blockIdx.y*256 + w*32, +32) -> 2 A fragments, JT n-tiles
//            of 16 -> 2*JT WMMA accumulators; B fragment reused twice.
//   A tiles move via GLOBAL_LOAD_ASYNC_TO_LDS_B128 (ASYNCcnt), double-buffered
//   in LDS one k-step ahead: tile k+1 DMA overlaps tile k's WMMAs. Async loads
//   complete in order and each tile is 4 ops, so s_wait_asynccnt 0x4 after
//   issuing tile k+1 guarantees tile k has landed.
// Template: RES -> add Cres f32; ACT_RELU -> relu; OUT_HALF -> f16 output;
//           JT  -> number of 16-wide N tiles per wave (grid.x = N/(16*JT)).
// ---------------------------------------------------------------------------
#define GBM 256
#define LDS_STRIDE 40   // halfs; 80 bytes -> keeps 16B alignment for b128 ops
#define LDS_TILE   (GBM * LDS_STRIDE)   // halfs per buffer

template <int RES, int ACT_RELU, int OUT_HALF, int JT>
__global__ __launch_bounds__(256) void gemm_wmma_f16(
    const _Float16* __restrict__ A, int lda,
    const _Float16* __restrict__ Bp, int ldb,
    const float* __restrict__ bias,
    const float* __restrict__ Cres,
    float* __restrict__ Df, _Float16* __restrict__ Dh,
    int ldd, int K)
{
  __shared__ _Float16 As[2 * LDS_TILE];   // 2 x 20 KB (double buffer)

  const int t    = threadIdx.x;
  const int wave = t >> 5;
  const int lane = t & 31;
  const int l16  = lane & 15;
  const int hi   = lane >> 4;          // 0: lanes 0-15, 1: lanes 16-31

  const int m0 = blockIdx.y * GBM;     // block row base
  const int mw = m0 + wave * 32;       // this wave's 32-row slab
  const int n0 = blockIdx.x * (16 * JT);

  v8f acc[2][JT] = {};

  // Async staging: thread t owns row t of the 256x32 tile (64 bytes).
  const _Float16* arow = A + (size_t)(m0 + t) * lda;
  const uint32_t lds_t = (uint32_t)(uintptr_t)(&As[t * LDS_STRIDE]);

  // each op copies 16 B from global vaddr+offset to LDS vdst+offset
  auto issue_tile = [&](int kk, int buf) {
    const unsigned long long ga = (unsigned long long)(uintptr_t)(arow + kk);
    const uint32_t la = lds_t + (uint32_t)buf * (LDS_TILE * 2);
    asm volatile(
        "global_load_async_to_lds_b128 %0, %1, off\n\t"
        "global_load_async_to_lds_b128 %0, %1, off offset:16\n\t"
        "global_load_async_to_lds_b128 %0, %1, off offset:32\n\t"
        "global_load_async_to_lds_b128 %0, %1, off offset:48"
        :: "v"(la), "v"(ga) : "memory");
  };

  const int nsteps = K >> 5;
  issue_tile(0, 0);                    // prologue: tile 0 -> buf 0

  for (int step = 0; step < nsteps; ++step) {
    const int cur = step & 1;
    if (step + 1 < nsteps) {
      // Safe: barrier at end of step-1 guarantees all reads of buf cur^1
      // (used in step-1) completed before this DMA overwrites it.
      issue_tile((step + 1) << 5, cur ^ 1);
      asm volatile("s_wait_asynccnt 0x4" ::: "memory");  // tile 'step' landed
    } else {
      asm volatile("s_wait_asynccnt 0x0" ::: "memory");
    }
    __syncthreads();                   // tile 'step' visible to all waves

    // A fragments (ISA 16-bit A layout): row = mw + s*16 + l16;
    // lanes 0-15: K = {0..7, 16..23}; lanes 16-31: K = {8..15, 24..31}.
    V16 a0, a1;
    {
      const _Float16* ap =
          &As[cur * LDS_TILE + ((wave * 32) + l16) * LDS_STRIDE + hi * 8];
      a0.h[0] = *(const v8h*)(ap);                           // ds_load_b128
      a0.h[1] = *(const v8h*)(ap + 16);
      const _Float16* ap1 = ap + 16 * LDS_STRIDE;
      a1.h[0] = *(const v8h*)(ap1);
      a1.h[1] = *(const v8h*)(ap1 + 16);
    }

    const int kk = step << 5;
#pragma unroll
    for (int jn = 0; jn < JT; ++jn) {
      const int n = n0 + jn * 16;
      // B fragment: col = n + l16; lanes 0-15 hold K={0..7,16..23}, lanes 16-31
      // hold K={8..15,24..31} (relative to kk), matching the A fragment.
      V16 b;
      const _Float16* bp = Bp + (size_t)(n + l16) * ldb + kk + hi * 8;
      b.h[0] = *(const v8h*)(bp);
      b.h[1] = *(const v8h*)(bp + 16);
      acc[0][jn] = __builtin_amdgcn_wmma_f32_16x16x32_f16(
          false, a0.v, false, b.v, (short)0, acc[0][jn], false, false);
      acc[1][jn] = __builtin_amdgcn_wmma_f32_16x16x32_f16(
          false, a1.v, false, b.v, (short)0, acc[1][jn], false, false);
    }
    __syncthreads();                   // all reads of buf 'cur' done
  }

  // Epilogue. C/D layout: VGPR r -> M = r + hi*8, N = l16.
#pragma unroll
  for (int s = 0; s < 2; ++s) {
#pragma unroll
    for (int jn = 0; jn < JT; ++jn) {
      const int col = n0 + jn * 16 + l16;
      const float bv = bias[col];
#pragma unroll
      for (int r = 0; r < 8; ++r) {
        const int row = mw + s * 16 + r + hi * 8;
        float val = acc[s][jn][r] + bv;
        if (RES)      val += Cres[(size_t)row * ldd + col];
        if (ACT_RELU) val = val > 0.0f ? val : 0.0f;
        if (OUT_HALF) Dh[(size_t)row * ldd + col] = (_Float16)val;
        else          Df[(size_t)row * ldd + col] = val;
      }
    }
  }
}

// ---------------------------------------------------------------------------
// Weight packing: W (K x N, f32) -> Wp (N x Kp, f16), zero pad K..Kp.
// ---------------------------------------------------------------------------
__global__ void pack_weight(const float* __restrict__ W, _Float16* __restrict__ Wp,
                            int K, int N, int Kp)
{
  int i = blockIdx.x * 256 + threadIdx.x;
  if (i >= N * Kp) return;
  int k = i % Kp, n = i / Kp;
  Wp[i] = (k < K) ? (_Float16)W[(size_t)k * N + n] : (_Float16)0.0f;
}

__global__ void zero_pad_ain(_Float16* __restrict__ A, int M)
{
  int i = blockIdx.x * 256 + threadIdx.x;
  if (i >= M * 16) return;
  int r = i >> 4, c = i & 15;
  A[(size_t)r * 224 + 208 + c] = (_Float16)0.0f;
}

// ---------------------------------------------------------------------------
// Channel LayerNorm of feature map (bs,C,H,W) -> f16 rows of A_in (M x 224).
// One wave per (b, pixel); lane covers channels c = lane, lane+32, ...
// ---------------------------------------------------------------------------
__global__ __launch_bounds__(256) void ln_feat(
    const float* __restrict__ feat, const float* __restrict__ g,
    const float* __restrict__ b, _Float16* __restrict__ out,
    int C, int colOff, int HW, int total_rows)
{
  int gw   = (int)((blockIdx.x * (size_t)blockDim.x + threadIdx.x) >> 5);
  int lane = threadIdx.x & 31;
  if (gw >= total_rows) return;
  int bi = gw / HW, q = gw % HW;
  const float* src = feat + (size_t)bi * C * HW + q;

  float v[4];
  float s = 0.0f;
  int cnt = 0;
  for (int c = lane; c < C; c += 32) { v[cnt] = src[(size_t)c * HW]; s += v[cnt]; ++cnt; }
#pragma unroll
  for (int off = 16; off; off >>= 1) s += __shfl_xor(s, off, 32);
  float mean = s / (float)C;

  float vs = 0.0f;
  cnt = 0;
  for (int c = lane; c < C; c += 32) { float d = v[cnt++] - mean; vs += d * d; }
#pragma unroll
  for (int off = 16; off; off >>= 1) vs += __shfl_xor(vs, off, 32);
  float rstd = rsqrtf(vs / (float)C + 1e-5f);

  _Float16* dst = out + (size_t)gw * 224 + colOff;
  cnt = 0;
  for (int c = lane; c < C; c += 32)
    dst[c] = (_Float16)((v[cnt++] - mean) * rstd * g[c] + b[c]);
}

// ---------------------------------------------------------------------------
// Row LayerNorm over e=256 (in-place on f32 Q), optional f16 copy.
// One wave per row, 8 elements per lane.
// ---------------------------------------------------------------------------
template <int WRITE_H>
__global__ __launch_bounds__(256) void ln_row256(
    float* __restrict__ Q, _Float16* __restrict__ QH,
    const float* __restrict__ g, const float* __restrict__ b, int M)
{
  int gw   = (int)((blockIdx.x * (size_t)blockDim.x + threadIdx.x) >> 5);
  int lane = threadIdx.x & 31;
  if (gw >= M) return;
  float* row = Q + (size_t)gw * 256;

  float v[8];
  float s = 0.0f;
#pragma unroll
  for (int i = 0; i < 8; ++i) { v[i] = row[lane + 32 * i]; s += v[i]; }
#pragma unroll
  for (int off = 16; off; off >>= 1) s += __shfl_xor(s, off, 32);
  float mean = s * (1.0f / 256.0f);

  float vs = 0.0f;
#pragma unroll
  for (int i = 0; i < 8; ++i) { float d = v[i] - mean; vs += d * d; }
#pragma unroll
  for (int off = 16; off; off >>= 1) vs += __shfl_xor(vs, off, 32);
  float rstd = rsqrtf(vs * (1.0f / 256.0f) + 1e-5f);

#pragma unroll
  for (int i = 0; i < 8; ++i) {
    int c = lane + 32 * i;
    float o = (v[i] - mean) * rstd * g[c] + b[c];
    row[c] = o;
    if (WRITE_H) QH[(size_t)gw * 256 + c] = (_Float16)o;
  }
}

// ---------------------------------------------------------------------------
// qp = f16(q + pos), pos[(y,x)] = [col_embed[x] | row_embed[y]].
// ---------------------------------------------------------------------------
__global__ void add_pos(const float* __restrict__ Q,
                        const float* __restrict__ row_embed,
                        const float* __restrict__ col_embed,
                        _Float16* __restrict__ QH, int Wdim, int nq, size_t total)
{
  size_t i = blockIdx.x * (size_t)256 + threadIdx.x;
  if (i >= total) return;
  int c    = (int)(i & 255);
  size_t m = i >> 8;
  int q = (int)(m % nq);
  int y = q / Wdim, x = q % Wdim;
  float p = (c < 128) ? col_embed[(size_t)x * 128 + c]
                      : row_embed[(size_t)y * 128 + (c - 128)];
  QH[i] = (_Float16)(Q[i] + p);
}

// ---------------------------------------------------------------------------
// Softmax over the 8 (NM*NP) logits per (b,q,head), in place.
// ---------------------------------------------------------------------------
__global__ void softmax8(float* __restrict__ AW, int total)
{
  int i = blockIdx.x * 256 + threadIdx.x;
  if (i >= total) return;
  float* p = AW + (size_t)i * 8;
  float mx = p[0];
#pragma unroll
  for (int j = 1; j < 8; ++j) mx = fmaxf(mx, p[j]);
  float e[8], s = 0.0f;
#pragma unroll
  for (int j = 0; j < 8; ++j) { e[j] = __expf(p[j] - mx); s += e[j]; }
  float inv = 1.0f / s;
#pragma unroll
  for (int j = 0; j < 8; ++j) p[j] = e[j] * inv;
}

// ---------------------------------------------------------------------------
// Deformable bilinear sampling + attention aggregation.
// One wave per (b,q,head); each lane handles 2 of the 64 head channels
// (one 4-byte v2h gather per corner). px = x + off.x (normalization cancels).
// ---------------------------------------------------------------------------
__global__ __launch_bounds__(256) void deform_sample(
    const _Float16* __restrict__ V1, const _Float16* __restrict__ V2,
    const float* __restrict__ OFF, const float* __restrict__ AW,
    _Float16* __restrict__ OUT, int Hh, int Ww, int nq, int total_waves)
{
  int gw   = (int)((blockIdx.x * (size_t)blockDim.x + threadIdx.x) >> 5);
  int lane = threadIdx.x & 31;
  if (gw >= total_waves) return;
  int h    = gw & 3;
  size_t m = (size_t)(gw >> 2);           // b*nq + q
  int q = (int)(m % nq);
  int y = q / Ww, x = q % Ww;
  int d0 = lane * 2;

  const float* offp = OFF + m * 64 + h * 16;   // [mod*8 + p*2 + xy]
  const float* awp  = AW  + m * 32 + h * 8;    // [mod*4 + p]
  size_t bbase = (m / nq) * (size_t)nq * 256;

  float o0 = 0.0f, o1 = 0.0f;
#pragma unroll
  for (int mod = 0; mod < 2; ++mod) {
    const _Float16* vb = (mod ? V2 : V1) + bbase + h * 64 + d0;
#pragma unroll
    for (int p = 0; p < 4; ++p) {
      float px = (float)x + offp[mod * 8 + p * 2 + 0];
      float py = (float)y + offp[mod * 8 + p * 2 + 1];
      float w  = awp[mod * 4 + p];
      float x0f = floorf(px), y0f = floorf(py);
      float wx = px - x0f, wy = py - y0f;
      int x0 = (int)x0f, y0 = (int)y0f;
#pragma unroll
      for (int cy = 0; cy < 2; ++cy) {
#pragma unroll
        for (int cx = 0; cx < 2; ++cx) {
          int xi = x0 + cx, yi = y0 + cy;
          float cw = (cx ? wx : 1.0f - wx) * (cy ? wy : 1.0f - wy) * w;
          if (xi >= 0 && xi < Ww && yi >= 0 && yi < Hh) {
            v2h pv = *(const v2h*)(vb + (size_t)(yi * Ww + xi) * 256);
            o0 += cw * (float)pv.x;
            o1 += cw * (float)pv.y;
          }
        }
      }
    }
  }
  _Float16* op = OUT + m * 256 + h * 64 + d0;
  v2h ov; ov.x = (_Float16)o0; ov.y = (_Float16)o1;
  *(v2h*)op = ov;
}

// ---------------------------------------------------------------------------
// Final transpose: out[b,c,y,x] = q[b, y*W+x, c].
// ---------------------------------------------------------------------------
__global__ void transpose_out(const float* __restrict__ Q, float* __restrict__ out,
                              int nq, int e, size_t total)
{
  size_t i = blockIdx.x * (size_t)256 + threadIdx.x;
  if (i >= total) return;
  size_t q = i % nq;
  size_t rest = i / nq;
  int c = (int)(rest % e);
  size_t b = rest / e;
  out[i] = Q[((size_t)b * nq + q) * e + c];
}

// ---------------------------------------------------------------------------
// Host orchestration.
// ---------------------------------------------------------------------------
static inline int cdiv(int a, int b) { return (a + b - 1) / b; }

extern "C" void kernel_launch(void* const* d_in, const int* in_sizes, int n_in,
                              void* d_out, int out_size, void* d_ws, size_t ws_size,
                              hipStream_t stream)
{
  (void)in_sizes; (void)n_in; (void)out_size; (void)ws_size;

  const int bs = 2, C1 = 80, C2 = 128, Hd = 128, Wd = 128;
  const int nq = Hd * Wd;                 // 16384
  const int M  = bs * nq;                 // 32768
  const int e  = 256, L = 6;

  const float* feat1     = (const float*)d_in[0];
  const float* feat2     = (const float*)d_in[1];
  const float* ln_img_g  = (const float*)d_in[2];
  const float* ln_img_b  = (const float*)d_in[3];
  const float* ln_pts_g  = (const float*)d_in[4];
  const float* ln_pts_b  = (const float*)d_in[5];
  const float* W_in      = (const float*)d_in[6];
  const float* b_in      = (const float*)d_in[7];
  const float* row_embed = (const float*)d_in[8];
  const float* col_embed = (const float*)d_in[9];
  const float* Wo   = (const float*)d_in[10]; const float* bo   = (const float*)d_in[11];
  const float* Wa   = (const float*)d_in[12]; const float* ba   = (const float*)d_in[13];
  const float* Wv1  = (const float*)d_in[14]; const float* bv1  = (const float*)d_in[15];
  const float* Wv2  = (const float*)d_in[16]; const float* bv2  = (const float*)d_in[17];
  const float* Wout = (const float*)d_in[18]; const float* bout = (const float*)d_in[19];
  const float* ln1g = (const float*)d_in[20]; const float* ln1b = (const float*)d_in[21];
  const float* Wf1  = (const float*)d_in[22]; const float* bf1  = (const float*)d_in[23];
  const float* Wf2  = (const float*)d_in[24]; const float* bf2  = (const float*)d_in[25];
  const float* ln2g = (const float*)d_in[26]; const float* ln2b = (const float*)d_in[27];

  // ---- workspace carve-out ----
  char* wsp = (char*)d_ws;
  auto alloc = [&](size_t bytes) -> char* {
    char* r = wsp;
    wsp += (bytes + 255) & ~(size_t)255;
    return r;
  };

  _Float16* A_IN = (_Float16*)alloc((size_t)M * 224 * 2); // [f1(80)|f2(128)|0(16)]
  float*    Q    = (float*)   alloc((size_t)M * 256 * 4);
  _Float16* QH   = (_Float16*)alloc((size_t)M * 256 * 2);
  _Float16* V1H  = (_Float16*)alloc((size_t)M * 256 * 2);
  _Float16* V2H  = (_Float16*)alloc((size_t)M * 256 * 2);
  _Float16* OUTH = (_Float16*)alloc((size_t)M * 256 * 2);
  _Float16* MIDH = (_Float16*)alloc((size_t)M * 512 * 2);
  float*    OFF  = (float*)   alloc((size_t)M * 64 * 4);
  float*    AW   = (float*)   alloc((size_t)M * 32 * 4);

  _Float16* W_inP = (_Float16*)alloc((size_t)256 * 224 * 2);
  _Float16 *WoP[6], *WaP[6], *Wv1P[6], *Wv2P[6], *WoutP[6], *Wf1P[6], *Wf2P[6];
  for (int l = 0; l < L; ++l) {
    WoP[l]   = (_Float16*)alloc((size_t)64  * 256 * 2);
    WaP[l]   = (_Float16*)alloc((size_t)32  * 256 * 2);
    Wv1P[l]  = (_Float16*)alloc((size_t)256 * 96  * 2);
    Wv2P[l]  = (_Float16*)alloc((size_t)256 * 128 * 2);
    WoutP[l] = (_Float16*)alloc((size_t)256 * 256 * 2);
    Wf1P[l]  = (_Float16*)alloc((size_t)512 * 256 * 2);
    Wf2P[l]  = (_Float16*)alloc((size_t)256 * 512 * 2);
  }

  auto pack = [&](const float* Wsrc, int K, int N, int Kp, _Float16* dst) {
    int tot = N * Kp;
    pack_weight<<<cdiv(tot, 256), 256, 0, stream>>>(Wsrc, dst, K, N, Kp);
  };

  // ---- pack all weights (once per call; deterministic) ----
  pack(W_in, 208, 256, 224, W_inP);
  for (int l = 0; l < L; ++l) {
    pack(Wo   + (size_t)l * 256 * 64,  256, 64,  256, WoP[l]);
    pack(Wa   + (size_t)l * 256 * 32,  256, 32,  256, WaP[l]);
    pack(Wv1  + (size_t)l * 80  * 256, 80,  256, 96,  Wv1P[l]);
    pack(Wv2  + (size_t)l * 128 * 256, 128, 256, 128, Wv2P[l]);
    pack(Wout + (size_t)l * 256 * 256, 256, 256, 256, WoutP[l]);
    pack(Wf1  + (size_t)l * 256 * 512, 256, 512, 256, Wf1P[l]);
    pack(Wf2  + (size_t)l * 512 * 256, 512, 256, 512, Wf2P[l]);
  }

  // ---- per-modality channel LN -> f16 concat buffer ----
  zero_pad_ain<<<cdiv(M * 16, 256), 256, 0, stream>>>(A_IN, M);
  {
    int blocks = cdiv(M * 32, 256);
    ln_feat<<<blocks, 256, 0, stream>>>(feat1, ln_img_g, ln_img_b, A_IN, C1, 0,  nq, M);
    ln_feat<<<blocks, 256, 0, stream>>>(feat2, ln_pts_g, ln_pts_b, A_IN, C2, 80, nq, M);
  }

  // ---- GEMM launcher ----
  // mode 0: plain f32 out (JT=4)   mode 1: f16 out (JT=4)
  // mode 2: residual f32 (JT=4)    mode 3: relu f16 (JT=4)
  // mode 4: plain f32 out (JT=2, for N=32)
  auto gemm = [&](int mode, const _Float16* A, int lda, const _Float16* Bp, int Kp,
                  const float* bias, const float* Cres, float* Df, _Float16* Dh,
                  int N, int K) {
    dim3 blk(256);
    dim3 g4(N / 64, M / GBM);
    dim3 g2(N / 32, M / GBM);
    switch (mode) {
      case 0: gemm_wmma_f16<0,0,0,4><<<g4, blk, 0, stream>>>(A, lda, Bp, Kp, bias, Cres, Df, Dh, N, K); break;
      case 1: gemm_wmma_f16<0,0,1,4><<<g4, blk, 0, stream>>>(A, lda, Bp, Kp, bias, Cres, Df, Dh, N, K); break;
      case 2: gemm_wmma_f16<1,0,0,4><<<g4, blk, 0, stream>>>(A, lda, Bp, Kp, bias, Cres, Df, Dh, N, K); break;
      case 3: gemm_wmma_f16<0,1,1,4><<<g4, blk, 0, stream>>>(A, lda, Bp, Kp, bias, Cres, Df, Dh, N, K); break;
      case 4: gemm_wmma_f16<0,0,0,2><<<g2, blk, 0, stream>>>(A, lda, Bp, Kp, bias, Cres, Df, Dh, N, K); break;
    }
  };

  // q0 = concat(f1,f2) @ W_in + b_in   (K padded to 224, zero rows in W_inP)
  gemm(0, A_IN, 224, W_inP, 224, b_in, nullptr, Q, nullptr, 256, 224);

  const size_t totQ = (size_t)M * 256;
  for (int l = 0; l < L; ++l) {
    // qp = f16(q + pos)
    add_pos<<<(int)((totQ + 255) / 256), 256, 0, stream>>>(Q, row_embed, col_embed, QH, Wd, nq, totQ);

    // sampling offsets and attention logits
    gemm(0, QH, 256, WoP[l], 256, bo + (size_t)l * 64, nullptr, OFF, nullptr, 64, 256);
    gemm(4, QH, 256, WaP[l], 256, ba + (size_t)l * 32, nullptr, AW,  nullptr, 32, 256);
    softmax8<<<cdiv(M * NH, 256), 256, 0, stream>>>(AW, M * NH);

    // value projections (f16 out); K padding handled by zeroed weight rows
    gemm(1, A_IN,      224, Wv1P[l], 96,  bv1 + (size_t)l * 256, nullptr, nullptr, V1H, 256, 96);
    gemm(1, A_IN + 80, 224, Wv2P[l], 128, bv2 + (size_t)l * 256, nullptr, nullptr, V2H, 256, 128);

    // deformable sampling + aggregation
    {
      int waves = M * NH;
      deform_sample<<<cdiv(waves * 32, 256), 256, 0, stream>>>(
          V1H, V2H, OFF, AW, OUTH, Hd, Wd, nq, waves);
    }

    // q += out @ Wout + bout ; LN1 (also emits f16 q)
    gemm(2, OUTH, 256, WoutP[l], 256, bout + (size_t)l * 256, Q, Q, nullptr, 256, 256);
    ln_row256<1><<<cdiv(M * 32, 256), 256, 0, stream>>>(Q, QH, ln1g + (size_t)l * 256, ln1b + (size_t)l * 256, M);

    // FFN: mid = relu(q @ Wf1 + bf1); q += mid @ Wf2 + bf2 ; LN2
    gemm(3, QH,   256, Wf1P[l], 256, bf1 + (size_t)l * 512, nullptr, nullptr, MIDH, 512, 256);
    gemm(2, MIDH, 512, Wf2P[l], 512, bf2 + (size_t)l * 256, Q, Q, nullptr, 256, 512);
    ln_row256<0><<<cdiv(M * 32, 256), 256, 0, stream>>>(Q, nullptr, ln2g + (size_t)l * 256, ln2b + (size_t)l * 256, M);
  }

  // out = transpose(q) -> (bs, e, H, W)
  {
    size_t tot = (size_t)bs * e * nq;
    transpose_out<<<(int)((tot + 255) / 256), 256, 0, stream>>>(Q, (float*)d_out, nq, e, tot);
  }
}
